// Refinement_33646773797080
// MI455X (gfx1250) — matrix operand order repaired and checked by reading, not compiled, not run on hardware
//
#include <hip/hip_runtime.h>
#include <hip/hip_bf16.h>

// ---------------------------------------------------------------------------
// Problem constants (from reference)
// ---------------------------------------------------------------------------
#define BATCH 128
#define NBOX  512
#define KSEL  6
#define FDIM  1024
#define H1DIM 512
#define H2DIM 256
#define M1    (BATCH * NBOX)   // 65536 rows, main MLP
#define M2    (BATCH * KSEL)   // 768 rows, refinement MLP
#define NMS_THRESH 0.5f
#define BN_EPS 1e-5f

typedef __attribute__((ext_vector_type(16))) __bf16        v16bf;
typedef __attribute__((ext_vector_type(8)))  float         v8f;
typedef __attribute__((ext_vector_type(8)))  unsigned int  v8u;

__device__ __forceinline__ unsigned short f2bf(float f) {
    unsigned u = __float_as_uint(f);
    unsigned r = u + 0x7FFFu + ((u >> 16) & 1u);   // round-to-nearest-even
    return (unsigned short)(r >> 16);
}
__device__ __forceinline__ float bf2f(unsigned short u) {
    return __uint_as_float(((unsigned)u) << 16);
}

// ---------------------------------------------------------------------------
// Async double-buffered bf16 WMMA GEMM.
//   C[M,N](bf16) = A[M,K](bf16) @ Wt[N,K](bf16, pre-transposed) + bias[N](f32)
// Block = 256 threads = 8 waves (4M x 2N). Block tile 128x128, K-step 32.
// Each wave computes a 32x64 output = 2x4 WMMA tiles (8 wmma / K-step).
// Tiles staged global->LDS with GLOBAL_LOAD_ASYNC_TO_LDS_B128 (ASYNCcnt),
// double buffered: async copy of step i+1 overlaps WMMA of step i.
// ---------------------------------------------------------------------------
#define BM 128
#define BN 128
#define BK 32
#define LDA 40   // ushort row stride: 80B, 16B-aligned chunks, conflict-free
#define LDB 40

__device__ __forceinline__ void issue_tile_async(
    const unsigned short* __restrict__ A, const unsigned short* __restrict__ Wt,
    int m0, int n0, int K, int k0, unsigned ldsA, unsigned ldsB, int tid)
{
    // A tile: BM rows x 64B of bf16 data per row; 4 x b128 chunks per row.
    #pragma unroll
    for (int s = 0; s < 2; ++s) {
        int c   = tid + s * 256;           // 0..511
        int row = c >> 2, cc = c & 3;
        unsigned gOff = ((unsigned)(m0 + row) * (unsigned)K + (unsigned)k0) * 2u
                        + (unsigned)cc * 16u;
        unsigned lds  = ldsA + (unsigned)row * (LDA * 2) + (unsigned)cc * 16u;
        asm volatile("global_load_async_to_lds_b128 %0, %1, %2"
                     :: "v"(lds), "v"(gOff), "s"(A) : "memory");
    }
    // B tile: BN rows x 64B per row; 4 x b128 chunks per row.
    #pragma unroll
    for (int s = 0; s < 2; ++s) {
        int c   = tid + s * 256;           // 0..511
        int row = c >> 2, cc = c & 3;
        unsigned gOff = ((unsigned)(n0 + row) * (unsigned)K + (unsigned)k0) * 2u
                        + (unsigned)cc * 16u;
        unsigned lds  = ldsB + (unsigned)row * (LDB * 2) + (unsigned)cc * 16u;
        asm volatile("global_load_async_to_lds_b128 %0, %1, %2"
                     :: "v"(lds), "v"(gOff), "s"(Wt) : "memory");
    }
}

__global__ __launch_bounds__(256) void gemm_bf16_wmma_async(
    const unsigned short* __restrict__ A,    // bf16 M x K
    const unsigned short* __restrict__ Wt,   // bf16 N x K (transposed weights)
    const float* __restrict__ bias,
    unsigned short* __restrict__ C,          // bf16 M x N
    int M, int N, int K)
{
    __shared__ __align__(16) unsigned short Asm[2][BM * LDA];
    __shared__ __align__(16) unsigned short Bsm[2][BN * LDB];

    const int tid  = threadIdx.x;
    const int lane = tid & 31;
    const int wave = tid >> 5;
    const int wm   = wave >> 1;    // 0..3  (32-row slice)
    const int wn   = wave & 1;     // 0..1  (64-col slice)
    const int half = lane >> 4;
    const int l15  = lane & 15;
    const int m0   = blockIdx.y * BM;
    const int n0   = blockIdx.x * BN;

    const unsigned ldsA0 = (unsigned)(size_t)(&Asm[0][0]);
    const unsigned ldsA1 = (unsigned)(size_t)(&Asm[1][0]);
    const unsigned ldsB0 = (unsigned)(size_t)(&Bsm[0][0]);
    const unsigned ldsB1 = (unsigned)(size_t)(&Bsm[1][0]);

    v8f acc[2][4] = {};

    const int nsteps = K / BK;
    issue_tile_async(A, Wt, m0, n0, K, 0, ldsA0, ldsB0, tid);

    for (int i = 0; i < nsteps; ++i) {
        const int cur = i & 1;
        if (i + 1 < nsteps) {
            issue_tile_async(A, Wt, m0, n0, K, (i + 1) * BK,
                             cur ? ldsA0 : ldsA1, cur ? ldsB0 : ldsB1, tid);
            // 4 outstanding async loads belong to step i+1; wait for step i's.
            asm volatile("s_wait_asynccnt 0x4" ::: "memory");
        } else {
            asm volatile("s_wait_asynccnt 0x0" ::: "memory");
        }
        __syncthreads();

        // ---- build fragments from LDS (ISA 7.12.2 layouts) ----
        union Frag { v16bf h; v8u u; };
        Frag afr[2], bfr[4];
        #pragma unroll
        for (int mi = 0; mi < 2; ++mi) {
            const unsigned short* arow = &Asm[cur][(wm * 32 + mi * 16 + l15) * LDA];
            #pragma unroll
            for (int v = 0; v < 8; ++v) {
                int k = (v < 4) ? (2 * v + 8 * half) : (16 + 2 * (v - 4) + 8 * half);
                afr[mi].u[v] = *(const unsigned int*)(arow + k);
            }
        }
        #pragma unroll
        for (int ni = 0; ni < 4; ++ni) {
            const unsigned short* brow = &Bsm[cur][(wn * 64 + ni * 16 + l15) * LDB];
            #pragma unroll
            for (int v = 0; v < 8; ++v)
                bfr[ni].u[v] = *(const unsigned int*)(brow + ((v < 4) ? (2 * v + 8 * half) : (16 + 2 * (v - 4) + 8 * half)));
        }
        #pragma unroll
        for (int mi = 0; mi < 2; ++mi)
            #pragma unroll
            for (int ni = 0; ni < 4; ++ni)
                acc[mi][ni] = __builtin_amdgcn_wmma_f32_16x16x32_bf16(
                    false, afr[mi].h, false, bfr[ni].h, (short)0,
                    acc[mi][ni], false, false);
        __syncthreads();
    }

    // D layout: VGPR v -> (M = v + 8*half, N = lane&15); bf16 store + bias
    #pragma unroll
    for (int mi = 0; mi < 2; ++mi)
        #pragma unroll
        for (int ni = 0; ni < 4; ++ni)
            #pragma unroll
            for (int v = 0; v < 8; ++v) {
                int m = m0 + wm * 32 + mi * 16 + v + 8 * half;
                int n = n0 + wn * 64 + ni * 16 + l15;
                C[(size_t)m * N + n] = f2bf(acc[mi][ni][v] + bias[n]);
            }
}

// ---------------------------------------------------------------------------
// f32 -> bf16 conversion (grid-stride)
// ---------------------------------------------------------------------------
__global__ void f32_to_bf16_kernel(const float* __restrict__ src,
                                   unsigned short* __restrict__ dst, size_t n)
{
    for (size_t i = (size_t)blockIdx.x * blockDim.x + threadIdx.x; i < n;
         i += (size_t)gridDim.x * blockDim.x)
        dst[i] = f2bf(src[i]);
}

// Wt[n*K+k] = bf16(W[k*N+n])  (write-coalesced transpose; weights are tiny)
__global__ void transpose_w_bf16(const float* __restrict__ W,
                                 unsigned short* __restrict__ Wt, int K, int N)
{
    int n = blockIdx.y;
    for (int k = blockIdx.x * 256 + threadIdx.x; k < K; k += gridDim.x * 256)
        Wt[(size_t)n * K + k] = f2bf(W[(size_t)k * N + n]);
}

// ---------------------------------------------------------------------------
// BatchNorm column stats over bf16 activations
// ---------------------------------------------------------------------------
__global__ __launch_bounds__(256) void bn_stats_bf16(
    const unsigned short* __restrict__ Y, int M, int N,
    float* __restrict__ mu, float* __restrict__ rinv)
{
    __shared__ float ssum[256], ssq[256];
    const int c32  = threadIdx.x & 31;
    const int rsub = threadIdx.x >> 5;
    const int col  = (blockIdx.x << 5) + c32;
    float s = 0.f, q = 0.f;
    for (int r = rsub; r < M; r += 8) {
        float y = bf2f(Y[(size_t)r * N + col]);
        s += y; q += y * y;
    }
    ssum[threadIdx.x] = s; ssq[threadIdx.x] = q;
    __syncthreads();
    if (rsub == 0) {
        for (int i = 1; i < 8; ++i) { s += ssum[i * 32 + c32]; q += ssq[i * 32 + c32]; }
        float mean = s / (float)M;
        float var  = q / (float)M - mean * mean;
        mu[col]   = mean;
        rinv[col] = rsqrtf(var + BN_EPS);
    }
}

__global__ void bn_relu_bf16(unsigned short* __restrict__ Y, int M, int N,
                             const float* __restrict__ mu, const float* __restrict__ rinv,
                             const float* __restrict__ g, const float* __restrict__ be)
{
    size_t total = (size_t)M * N;
    for (size_t i = (size_t)blockIdx.x * blockDim.x + threadIdx.x; i < total;
         i += (size_t)gridDim.x * blockDim.x) {
        int c = (int)(i % (size_t)N);
        float h = g[c] * (bf2f(Y[i]) - mu[c]) * rinv[c] + be[c];
        Y[i] = f2bf(h > 0.f ? h : 0.f);
    }
}

// ---------------------------------------------------------------------------
// 2-class head + softmax (pred_labels), bf16 activations
// ---------------------------------------------------------------------------
__global__ void head2_softmax(const unsigned short* __restrict__ H,
                              const float* __restrict__ W, const float* __restrict__ b,
                              float* __restrict__ out, int M, int Kd)
{
    int r = blockIdx.x * blockDim.x + threadIdx.x;
    if (r >= M) return;
    const unsigned short* h = H + (size_t)r * Kd;
    float l0 = b[0], l1 = b[1];
    for (int k = 0; k < Kd; ++k) {
        float hv = bf2f(h[k]);
        l0 += hv * W[k * 2 + 0];
        l1 += hv * W[k * 2 + 1];
    }
    float mx = fmaxf(l0, l1);
    float e0 = __expf(l0 - mx), e1 = __expf(l1 - mx);
    float inv = 1.f / (e0 + e1);
    out[(size_t)r * 2 + 0] = e0 * inv;
    out[(size_t)r * 2 + 1] = e1 * inv;
}

// ---------------------------------------------------------------------------
// Refinement heads: out7 = H@rw3+rb3, conf = softmax(H@rw4+rb4)
// ---------------------------------------------------------------------------
__global__ void refine_head(const unsigned short* __restrict__ H,
                            const float* __restrict__ W3, const float* __restrict__ b3,
                            const float* __restrict__ W4, const float* __restrict__ b4,
                            float* __restrict__ out7, float* __restrict__ conf,
                            int M, int Kd)
{
    int r = blockIdx.x * blockDim.x + threadIdx.x;
    if (r >= M) return;
    const unsigned short* h = H + (size_t)r * Kd;
    float o[7];
    #pragma unroll
    for (int j = 0; j < 7; ++j) o[j] = b3[j];
    float c0 = b4[0], c1 = b4[1];
    for (int k = 0; k < Kd; ++k) {
        float hv = bf2f(h[k]);
        #pragma unroll
        for (int j = 0; j < 7; ++j) o[j] += hv * W3[k * 7 + j];
        c0 += hv * W4[k * 2 + 0];
        c1 += hv * W4[k * 2 + 1];
    }
    #pragma unroll
    for (int j = 0; j < 7; ++j) out7[(size_t)r * 7 + j] = o[j];
    float mx = fmaxf(c0, c1);
    float e0 = __expf(c0 - mx), e1 = __expf(c1 - mx);
    float inv = 1.f / (e0 + e1);
    conf[(size_t)r * 2 + 0] = e0 * inv;
    conf[(size_t)r * 2 + 1] = e1 * inv;
}

// ---------------------------------------------------------------------------
// Per-batch NMS: bitonic sort by -score in LDS, sequential suppression scan
// (lax.scan semantics), stable compaction. One block (8 waves) per batch.
// ---------------------------------------------------------------------------
__global__ __launch_bounds__(256) void nms_kernel(
    const float* __restrict__ boxes,    // (B, NBOX, 7)
    const float* __restrict__ scores,   // (B, NBOX)
    int* __restrict__ nms_idx)          // (B, KSEL)
{
    __shared__ float skey[NBOX];
    __shared__ int   sidx[NBOX];
    __shared__ float bx1[NBOX], by1[NBOX], bx2[NBOX], by2[NBOX], sarea[NBOX];
    __shared__ int   supp[NBOX], keepf[NBOX], comp[NBOX];

    const int b = blockIdx.x;
    const int t = threadIdx.x;
    const float* bx = boxes + (size_t)b * NBOX * 7;

    for (int i = t; i < NBOX; i += 256) {
        float x = bx[i * 7 + 0], z = bx[i * 7 + 2];
        float w = bx[i * 7 + 4], l = bx[i * 7 + 5];
        float x1 = x - l * 0.5f, y1 = z - w * 0.5f;
        float x2 = x + l * 0.5f, y2 = z + w * 0.5f;
        bx1[i] = x1; by1[i] = y1; bx2[i] = x2; by2[i] = y2;
        sarea[i] = (x2 - x1) * (y2 - y1);
        skey[i] = scores[(size_t)b * NBOX + i];
        sidx[i] = i;
        supp[i] = 0;
    }
    __syncthreads();

    for (int k = 2; k <= NBOX; k <<= 1) {
        for (int j = k >> 1; j > 0; j >>= 1) {
            for (int i = t; i < NBOX; i += 256) {
                int p = i ^ j;
                if (p > i) {
                    bool dir = ((i & k) == 0);
                    if ((skey[i] < skey[p]) == dir) {
                        float tk = skey[i]; skey[i] = skey[p]; skey[p] = tk;
                        int   ti = sidx[i]; sidx[i] = sidx[p]; sidx[p] = ti;
                    }
                }
            }
            __syncthreads();
        }
    }

    for (int i = 0; i < NBOX; ++i) {
        if (t == 0) keepf[i] = supp[i] ? 0 : 1;
        __syncthreads();
        if (keepf[i]) {
            int oi = sidx[i];
            float a1 = bx1[oi], a2 = by1[oi], a3 = bx2[oi], a4 = by2[oi], aa = sarea[oi];
            for (int j2 = t; j2 < NBOX; j2 += 256) {
                int oj = sidx[j2];
                float ix1 = fmaxf(a1, bx1[oj]);
                float iy1 = fmaxf(a2, by1[oj]);
                float ix2 = fminf(a3, bx2[oj]);
                float iy2 = fminf(a4, by2[oj]);
                float inter = fmaxf(ix2 - ix1, 0.f) * fmaxf(iy2 - iy1, 0.f);
                float iou = inter / (aa + sarea[oj] - inter + 1e-8f);
                if (iou > NMS_THRESH) supp[j2] = 1;
            }
        }
        __syncthreads();
    }

    if (t == 0) {
        int m = 0;
        for (int p = 0; p < NBOX; ++p) if (keepf[p])  comp[m++]  = sidx[p];
        int mm = m;
        for (int p = 0; p < NBOX; ++p) if (!keepf[p]) comp[mm++] = sidx[p];
        for (int q = 0; q < KSEL; ++q)
            nms_idx[b * KSEL + q] = (m > 0) ? comp[q % m] : (NBOX - KSEL + q);
    }
}

// ---------------------------------------------------------------------------
// Gather selected rows: bf16 features -> workspace; boxes/gt/lbl -> d_out
// ---------------------------------------------------------------------------
__global__ __launch_bounds__(256) void gather_kernel(
    const unsigned short* __restrict__ featBf, const float* __restrict__ boxes,
    const int* __restrict__ gt, const float* __restrict__ lbl,
    const int* __restrict__ nms_idx,
    unsigned short* __restrict__ featsSel, float* __restrict__ boxesOut,
    float* __restrict__ gtOut, float* __restrict__ lblOut)
{
    int r = blockIdx.x;              // 0..M2-1
    int b = r / KSEL;
    int src = nms_idx[r];
    const unsigned short* fsrc = featBf + ((size_t)b * NBOX + src) * FDIM;
    unsigned short* fdst = featsSel + (size_t)r * FDIM;
    for (int i = threadIdx.x; i < FDIM; i += 256) fdst[i] = fsrc[i];
    if (threadIdx.x < 7)
        boxesOut[r * 7 + threadIdx.x] = boxes[((size_t)b * NBOX + src) * 7 + threadIdx.x];
    if (threadIdx.x == 0) {
        gtOut[r]  = (float)gt[(size_t)b * NBOX + src];
        lblOut[r] = lbl[(size_t)b * NBOX + src];
    }
}

__global__ void copy_f32(const float* __restrict__ src, float* __restrict__ dst, int n)
{
    int i = blockIdx.x * blockDim.x + threadIdx.x;
    if (i < n) dst[i] = src[i];
}

// ---------------------------------------------------------------------------
// Host-side orchestration
// ---------------------------------------------------------------------------
extern "C" void kernel_launch(void* const* d_in, const int* in_sizes, int n_in,
                              void* d_out, int out_size, void* d_ws, size_t ws_size,
                              hipStream_t stream)
{
    (void)in_sizes; (void)n_in; (void)out_size; (void)ws_size;

    const float* features = (const float*)d_in[0];
    const float* boxes    = (const float*)d_in[1];
    const float* iou_sc   = (const float*)d_in[2];
    const float* lbl_bin  = (const float*)d_in[3];
    const int*   gt_idx   = (const int*)  d_in[4];
    const float* w1  = (const float*)d_in[5];   const float* b1  = (const float*)d_in[6];
    const float* w2  = (const float*)d_in[7];   const float* b2  = (const float*)d_in[8];
    const float* w3  = (const float*)d_in[9];   const float* b3  = (const float*)d_in[10];
    const float* g1  = (const float*)d_in[11];  const float* be1 = (const float*)d_in[12];
    const float* g2  = (const float*)d_in[13];  const float* be2 = (const float*)d_in[14];
    const float* rw1 = (const float*)d_in[15];  const float* rb1 = (const float*)d_in[16];
    const float* rw2 = (const float*)d_in[17];  const float* rb2 = (const float*)d_in[18];
    const float* rw3 = (const float*)d_in[19];  const float* rb3 = (const float*)d_in[20];
    const float* rw4 = (const float*)d_in[21];  const float* rb4 = (const float*)d_in[22];
    const float* rg1 = (const float*)d_in[23];  const float* rbe1= (const float*)d_in[24];
    const float* rg2 = (const float*)d_in[25];  const float* rbe2= (const float*)d_in[26];

    float* out = (float*)d_out;
    float* o_out7   = out + 0;                               // (B,K,7)
    float* o_conf   = o_out7 + (size_t)M2 * 7;               // (768,2)
    float* o_boxes  = o_conf + (size_t)M2 * 2;               // (B,K,7)
    float* o_gt     = o_boxes + (size_t)M2 * 7;              // (B,K)
    float* o_pred   = o_gt + M2;                             // (B,N,2)
    float* o_lblcp  = o_pred + (size_t)M1 * 2;               // (B,N)
    float* o_lblsel = o_lblcp + M1;                          // (B,K)

    // workspace: bf16 buffers first, then f32 stats, then ints
    unsigned short* p = (unsigned short*)d_ws;
    unsigned short* featBf = p; p += (size_t)M1 * FDIM;
    unsigned short* Y1bf   = p; p += (size_t)M1 * H1DIM;
    unsigned short* Y2bf   = p; p += (size_t)M1 * H2DIM;
    unsigned short* FSbf   = p; p += (size_t)M2 * FDIM;
    unsigned short* R1bf   = p; p += (size_t)M2 * H1DIM;
    unsigned short* R2bf   = p; p += (size_t)M2 * H2DIM;
    unsigned short* w1t    = p; p += (size_t)FDIM * H1DIM;
    unsigned short* w2t    = p; p += (size_t)H1DIM * H2DIM;
    unsigned short* rw1t   = p; p += (size_t)FDIM * H1DIM;
    unsigned short* rw2t   = p; p += (size_t)H1DIM * H2DIM;
    float* fp   = (float*)p;
    float* mu1  = fp;           float* ri1  = fp + H1DIM;
    float* mu2  = ri1 + H1DIM;  float* ri2  = mu2 + H2DIM;
    float* mur1 = ri2 + H2DIM;  float* rir1 = mur1 + H1DIM;
    float* mur2 = rir1 + H1DIM; float* rir2 = mur2 + H2DIM;
    int*   nmsI = (int*)(rir2 + H2DIM);

    // ---- one-time precision conversion / weight transpose ----
    f32_to_bf16_kernel<<<8192, 256, 0, stream>>>(features, featBf, (size_t)M1 * FDIM);
    transpose_w_bf16<<<dim3(FDIM / 256, H1DIM),  256, 0, stream>>>(w1,  w1t,  FDIM,  H1DIM);
    transpose_w_bf16<<<dim3(H1DIM / 256, H2DIM), 256, 0, stream>>>(w2,  w2t,  H1DIM, H2DIM);
    transpose_w_bf16<<<dim3(FDIM / 256, H1DIM),  256, 0, stream>>>(rw1, rw1t, FDIM,  H1DIM);
    transpose_w_bf16<<<dim3(H1DIM / 256, H2DIM), 256, 0, stream>>>(rw2, rw2t, H1DIM, H2DIM);

    // ---- classification MLP (async WMMA path) ----
    gemm_bf16_wmma_async<<<dim3(H1DIM / BN, M1 / BM), 256, 0, stream>>>(
        featBf, w1t, b1, Y1bf, M1, H1DIM, FDIM);
    bn_stats_bf16<<<H1DIM / 32, 256, 0, stream>>>(Y1bf, M1, H1DIM, mu1, ri1);
    bn_relu_bf16<<<4096, 256, 0, stream>>>(Y1bf, M1, H1DIM, mu1, ri1, g1, be1);

    gemm_bf16_wmma_async<<<dim3(H2DIM / BN, M1 / BM), 256, 0, stream>>>(
        Y1bf, w2t, b2, Y2bf, M1, H2DIM, H1DIM);
    bn_stats_bf16<<<H2DIM / 32, 256, 0, stream>>>(Y2bf, M1, H2DIM, mu2, ri2);
    bn_relu_bf16<<<4096, 256, 0, stream>>>(Y2bf, M1, H2DIM, mu2, ri2, g2, be2);

    head2_softmax<<<M1 / 256, 256, 0, stream>>>(Y2bf, w3, b3, o_pred, M1, H2DIM);
    copy_f32<<<M1 / 256, 256, 0, stream>>>(lbl_bin, o_lblcp, M1);

    // ---- NMS + gather ----
    nms_kernel<<<BATCH, 256, 0, stream>>>(boxes, iou_sc, nmsI);
    gather_kernel<<<M2, 256, 0, stream>>>(featBf, boxes, gt_idx, lbl_bin, nmsI,
                                          FSbf, o_boxes, o_gt, o_lblsel);

    // ---- refinement MLP (async WMMA path) ----
    gemm_bf16_wmma_async<<<dim3(H1DIM / BN, M2 / BM), 256, 0, stream>>>(
        FSbf, rw1t, rb1, R1bf, M2, H1DIM, FDIM);
    bn_stats_bf16<<<H1DIM / 32, 256, 0, stream>>>(R1bf, M2, H1DIM, mur1, rir1);
    bn_relu_bf16<<<512, 256, 0, stream>>>(R1bf, M2, H1DIM, mur1, rir1, rg1, rbe1);

    gemm_bf16_wmma_async<<<dim3(H2DIM / BN, M2 / BM), 256, 0, stream>>>(
        R1bf, rw2t, rb2, R2bf, M2, H2DIM, H1DIM);
    bn_stats_bf16<<<H2DIM / 32, 256, 0, stream>>>(R2bf, M2, H2DIM, mur2, rir2);
    bn_relu_bf16<<<256, 256, 0, stream>>>(R2bf, M2, H2DIM, mur2, rir2, rg2, rbe2);

    refine_head<<<(M2 + 255) / 256, 256, 0, stream>>>(
        R2bf, rw3, rb3, rw4, rb4, o_out7, o_conf, M2, H2DIM);
}
